// Attention_17557826306415
// MI455X (gfx1250) — matrix operand order repaired and checked by reading, not compiled, not run on hardware
//
#include <hip/hip_runtime.h>

typedef __attribute__((ext_vector_type(16))) _Float16 v16h;
typedef __attribute__((ext_vector_type(8)))  float    v8f;
typedef __attribute__((ext_vector_type(4)))  float    v4f;   // native vec for NT builtins

#define B_ 4
#define H_ 16
#define S_ 1024
#define D_ 64
#define NWAVE 8

__global__ __launch_bounds__(256) void topk_attn_kernel(
    const float* __restrict__ Q,
    const float* __restrict__ K,
    const float* __restrict__ V,
    const int*   __restrict__ Msk,
    float* __restrict__ Out,     // (B,H,S,D)
    float* __restrict__ P)       // (B,H,S,S)
{
    extern __shared__ float sS[];            // 16 x 1024 fp32 scores/probs = 64 KB

    const int tid  = threadIdx.x;
    const int lane = tid & 31;
    const int wave = tid >> 5;
    const int n16  = lane & 15;
    const int half = lane >> 4;

    const int wg = blockIdx.x;
    const int qt = wg & 63;                  // query tile (0..63)
    const int h  = (wg >> 6) & 15;
    const int b  = wg >> 10;

    const float* Qb = Q + (((size_t)(b*H_ + h))*S_ + (size_t)qt*16) * D_;
    const float* Kb = K + ((size_t)(b*H_ + h))*S_*D_;
    const float* Vb = V + ((size_t)(b*H_ + h))*S_*D_;
    const int*   Mb = Msk + ((size_t)b)*S_*S_ + (size_t)(qt*16)*S_;
    float* Pb = P + ((size_t)(b*H_ + h))*S_*S_ + (size_t)(qt*16)*S_;
    float* Ob = Out + (((size_t)(b*H_ + h))*S_ + (size_t)qt*16)*D_;

    // ---------------- Stage 1: S = exp(Q K^T) / sqrt(D) ----------------
    // A fragments: Q tile 16x32 halves (d=0..31 and d=32..63), per ISA 16-bit A layout.
    v16h a0, a1;
    {
        const float* qr = Qb + (size_t)n16 * D_;
        #pragma unroll
        for (int j = 0; j < 8; ++j) {
            int k0 = ((j < 4) ? 2*j : 16 + 2*(j - 4)) + 8*half;
            a0[2*j]   = (_Float16)qr[k0];
            a0[2*j+1] = (_Float16)qr[k0 + 1];
            a1[2*j]   = (_Float16)qr[32 + k0];
            a1[2*j+1] = (_Float16)qr[32 + k0 + 1];
        }
    }

    for (int jt = wave; jt < S_/16; jt += NWAVE) {
        const int kb = jt * 16;
        // B fragments: B[row=d][col=key] = K[key][d]; lane's key = kb+n16.
        v16h b0, b1;
        const float* kr = Kb + (size_t)(kb + n16) * D_;
        #pragma unroll
        for (int j = 0; j < 8; ++j) {
            int k0 = ((j < 4) ? 2*j : 16 + 2*(j - 4)) + 8*half;
            b0[2*j]   = (_Float16)kr[k0];
            b0[2*j+1] = (_Float16)kr[k0 + 1];
            b1[2*j]   = (_Float16)kr[32 + k0];
            b1[2*j+1] = (_Float16)kr[32 + k0 + 1];
        }
        v8f c = {};
        c = __builtin_amdgcn_wmma_f32_16x16x32_f16(false, a0, false, b0, (short)0, c, false, false);
        c = __builtin_amdgcn_wmma_f32_16x16x32_f16(false, a1, false, b1, (short)0, c, false, false);
        #pragma unroll
        for (int r = 0; r < 8; ++r) {
            int m = r + 8*half;                       // C layout: VGPR r -> row r / r+8
            sS[m*S_ + kb + n16] = __expf(c[r]) * 0.125f;
        }
    }
    __syncthreads();

    // ---------------- Stage 2: top-16 threshold + masked softmax ----------------
    {
        const int row = 2*wave + half;               // each half-wave owns one row
        float v[64];
        #pragma unroll
        for (int i = 0; i < 64; ++i) v[i] = sS[row*S_ + n16 + 16*i];

        float thr = 3.0e38f, rowmax = -3.0e38f;
        for (int it = 0; it < 16; ++it) {            // iterative k-th max
            float m = -3.0e38f;
            #pragma unroll
            for (int i = 0; i < 64; ++i) {
                float x = v[i];
                m = (x < thr && x > m) ? x : m;
            }
            #pragma unroll
            for (int off = 1; off < 16; off <<= 1)
                m = fmaxf(m, __shfl_xor(m, off, 32));
            if (it == 0) rowmax = m;
            thr = m;
        }

        // mask is a read-once 256 MB stream -> non-temporal loads (don't thrash L2)
        const int* mrow = Mb + (size_t)row * S_;
        float denom = 0.f;
        #pragma unroll
        for (int i = 0; i < 64; ++i) {
            int col = n16 + 16*i;
            int mv = __builtin_nontemporal_load(&mrow[col]);
            bool keep = (v[i] >= thr) && (mv != 0);
            float z = keep ? __expf(v[i] - rowmax) : 0.f;
            v[i] = z;
            denom += z;
        }
        #pragma unroll
        for (int off = 1; off < 16; off <<= 1)
            denom += __shfl_xor(denom, off, 32);
        float inv = 1.0f / denom;
        #pragma unroll
        for (int i = 0; i < 64; ++i)
            sS[row*S_ + n16 + 16*i] = v[i] * inv;
    }
    __syncthreads();

    // ------- Stage 2b: coalesced p_attn store (dominant HBM traffic) -------
    // 268 MB written once, never re-read on device -> non-temporal stores so
    // the 192 MB L2 keeps K/V (reused by 64 WGs per head) instead.
    {
        const v4f* src = (const v4f*)sS;
        v4f* dst = (v4f*)Pb;
        #pragma unroll
        for (int i = 0; i < 16; ++i) {
            v4f val = src[i*256 + tid];
            __builtin_nontemporal_store(val, &dst[i*256 + tid]);
        }
    }

    // ---------------- Stage 3: Out = P @ V ----------------
    {
        const int nt = wave & 3;                     // output d-tile (0..3)
        const int kh = wave >> 2;                    // key half (0..1)
        const int dbase = nt * 16;
        v8f c = {};
        for (int kb = kh*512; kb < kh*512 + 512; kb += 32) {
            v16h a, bv;
            #pragma unroll
            for (int j = 0; j < 8; ++j) {            // A = P rows, 16-bit A layout
                int k0 = ((j < 4) ? 2*j : 16 + 2*(j - 4)) + 8*half;
                a[2*j]   = (_Float16)sS[n16*S_ + kb + k0];
                a[2*j+1] = (_Float16)sS[n16*S_ + kb + k0 + 1];
            }
            #pragma unroll
            for (int j = 0; j < 8; ++j) {            // B[row=key][col=d] = V[key][d]
                int r0 = ((j < 4) ? 2*j : 16 + 2*(j - 4)) + 8*half;
                bv[2*j]   = (_Float16)Vb[(size_t)(kb + r0    )*D_ + dbase + n16];
                bv[2*j+1] = (_Float16)Vb[(size_t)(kb + r0 + 1)*D_ + dbase + n16];
            }
            c = __builtin_amdgcn_wmma_f32_16x16x32_f16(false, a, false, bv, (short)0, c, false, false);
        }
        __syncthreads();                              // all P reads done; reuse LDS
        #pragma unroll
        for (int r = 0; r < 8; ++r)
            sS[wave*256 + r*32 + lane] = c[r];
        __syncthreads();
        if (wave < 4) {                               // combine the two key-halves
            #pragma unroll
            for (int r = 0; r < 8; ++r) {
                float sum = sS[wave*256 + r*32 + lane] + sS[(wave + 4)*256 + r*32 + lane];
                int m = r + 8*half;
                Ob[(size_t)m*D_ + wave*16 + n16] = sum;
            }
        }
    }
}

extern "C" void kernel_launch(void* const* d_in, const int* in_sizes, int n_in,
                              void* d_out, int out_size, void* d_ws, size_t ws_size,
                              hipStream_t stream) {
    const float* Q = (const float*)d_in[0];
    const float* K = (const float*)d_in[1];
    const float* V = (const float*)d_in[2];
    const int*   M = (const int*)d_in[3];
    // d_in[4] is k (==16), baked into the kernel's top-k loop.
    float* Out = (float*)d_out;
    float* P   = Out + (size_t)B_ * H_ * S_ * D_;

    dim3 grid(B_ * H_ * (S_ / 16));   // 4096 workgroups, one 16-row query tile each
    dim3 block(256);                  // 8 wave32 waves
    topk_attn_kernel<<<grid, block, 64 * 1024, stream>>>(Q, K, V, M, Out, P);
}
